// SpectralAttention_80358838108582
// MI455X (gfx1250) — matrix operand order repaired and checked by reading, not compiled, not run on hardware
//
#include <hip/hip_runtime.h>

// SpectralAttention for MI455X (gfx1250, wave32, WMMA + TDM).
// B=4, C=192, H=W=64, HW=4096. Outputs: out [B,C,H,W] fp32 then corr [B,HW,HW] fp32.

#define B_DIM   4
#define C_DIM   192
#define HW_DIM  4096
#define NSPLIT  16
#define TPB     256
#define WAVES_PER_BLOCK 8

#if defined(__has_builtin)
#if __has_builtin(__builtin_amdgcn_tensor_load_to_lds)
#define HAVE_TDM 1
#endif
#if __has_builtin(__builtin_amdgcn_cvt_pk_bf16_f32)
#define HAVE_CVT_PK_BF16 1
#endif
#endif
#ifndef HAVE_TDM
#define HAVE_TDM 0
#endif
#ifndef HAVE_CVT_PK_BF16
#define HAVE_CVT_PK_BF16 0
#endif

typedef __attribute__((ext_vector_type(16))) __bf16   v16bf;
typedef __attribute__((ext_vector_type(8)))  __bf16   v8bf;
typedef __attribute__((ext_vector_type(2)))  __bf16   v2bf;
typedef __attribute__((ext_vector_type(8)))  float    v8f;
typedef __attribute__((ext_vector_type(4)))  float    f32x4;
typedef __attribute__((ext_vector_type(4)))  unsigned u32x4;
typedef __attribute__((ext_vector_type(8)))  unsigned u32x8;
typedef __attribute__((ext_vector_type(8)))  int      i32x8;
typedef __attribute__((ext_vector_type(4)))  int      i32x4;

// ---------------- scalar helpers ----------------

// two fp32 -> packed bf16 pair (result = bf16(b)<<16 | bf16(a))
__device__ __forceinline__ unsigned pack2bf(float a, float b) {
#if HAVE_CVT_PK_BF16
  v2bf r = __builtin_amdgcn_cvt_pk_bf16_f32(a, b);     // v_cvt_pk_bf16_f32 (HW RNE)
  return __builtin_bit_cast(unsigned, r);
#else
  unsigned ua = __float_as_uint(a); ua += 0x7FFFu + ((ua >> 16) & 1u);
  unsigned ub = __float_as_uint(b); ub += 0x7FFFu + ((ub >> 16) & 1u);
  return __builtin_amdgcn_perm(ub, ua, 0x07060302u);   // RNE + v_perm pack
#endif
}
__device__ __forceinline__ unsigned short f2bf(float f) {
#if HAVE_CVT_PK_BF16
  return (unsigned short)pack2bf(f, 0.0f);
#else
  unsigned int u = __float_as_uint(f);
  u += 0x7FFFu + ((u >> 16) & 1u);
  return (unsigned short)(u >> 16);
#endif
}
__device__ __forceinline__ __bf16 bfraw(unsigned short u) {
  return __builtin_bit_cast(__bf16, u);
}

__device__ __forceinline__ void atomicMaxF(float* addr, float val) {
  unsigned int* ua = (unsigned int*)addr;
  unsigned int cur = *ua;
  while (__uint_as_float(cur) < val) {
    unsigned int prev = atomicCAS(ua, cur, __float_as_uint(val));
    if (prev == cur) break;
    cur = prev;
  }
}

__device__ __forceinline__ v8f wmma_bf16(v16bf a, v16bf b, v8f c) {
  return __builtin_amdgcn_wmma_f32_16x16x32_bf16(false, a, false, b, (short)0, c,
                                                 false, false);
}

// ---------------- WMMA fragment loaders (CDNA5 ISA 7.12.2 layouts) ----------------
// A (16x32 MxK): lane m = lane&15; klo=(lane>=16)?8:0; elems 0..7 -> K=klo+e,
//                elems 8..15 -> K=16+klo+e.
// B (32x16 KxN): lane n = lane&15; same K mapping as A (klo+e, 16+klo+e).

__device__ __forceinline__ v16bf cat8(v8bf lo, v8bf hi) {
  return __builtin_shufflevector(lo, hi, 0, 1, 2, 3, 4, 5, 6, 7, 8, 9, 10, 11, 12, 13,
                                 14, 15);
}

// A fragment from row-major bf16 [M, K] (two b128 loads per lane)
__device__ __forceinline__ v16bf loadA_bf16v(const unsigned short* __restrict__ base,
                                             int row0, int col0, int ld, int lane) {
  const unsigned short* r =
      base + (size_t)(row0 + (lane & 15)) * ld + col0 + ((lane >> 4) << 3);
  v8bf lo = __builtin_bit_cast(v8bf, *(const u32x4*)r);
  v8bf hi = __builtin_bit_cast(v8bf, *(const u32x4*)(r + 16));
  return cat8(lo, hi);
}

// A fragment from row-major fp32 [M, K]; convert via packed bf16 conversion
__device__ __forceinline__ v16bf loadA_f32v(const float* __restrict__ base,
                                            int row0, int col0, int ld, int lane) {
  const float* r = base + (size_t)(row0 + (lane & 15)) * ld + col0 + ((lane >> 4) << 3);
  f32x4 f0 = *(const f32x4*)(r);
  f32x4 f1 = *(const f32x4*)(r + 4);
  f32x4 f2 = *(const f32x4*)(r + 16);
  f32x4 f3 = *(const f32x4*)(r + 20);
  u32x8 w;
  w[0] = pack2bf(f0[0], f0[1]); w[1] = pack2bf(f0[2], f0[3]);
  w[2] = pack2bf(f1[0], f1[1]); w[3] = pack2bf(f1[2], f1[3]);
  w[4] = pack2bf(f2[0], f2[1]); w[5] = pack2bf(f2[2], f2[3]);
  w[6] = pack2bf(f3[0], f3[1]); w[7] = pack2bf(f3[2], f3[3]);
  return __builtin_bit_cast(v16bf, w);
}

// B fragment gathered from row-major bf16 [N, K] storage (B^T in memory, contiguous)
__device__ __forceinline__ v16bf loadBT_bf16v(const unsigned short* __restrict__ base,
                                              int row0, int col0, int ld, int lane) {
  const unsigned short* p =
      base + (size_t)(col0 + (lane & 15)) * ld + row0 + ((lane >> 4) << 3);
  v8bf lo = __builtin_bit_cast(v8bf, *(const u32x4*)p);
  v8bf hi = __builtin_bit_cast(v8bf, *(const u32x4*)(p + 16));
  return cat8(lo, hi);
}

// B fragment from an LDS-staged [64 n][32 k] bf16 panel (two ds_load_b128 per lane)
__device__ __forceinline__ v16bf loadB_lds(const unsigned short* tile, int t, int lane) {
  const unsigned short* p = tile + (((t << 4) + (lane & 15)) << 5) + ((lane >> 4) << 3);
  v8bf lo = __builtin_bit_cast(v8bf, *(const u32x4*)p);
  v8bf hi = __builtin_bit_cast(v8bf, *(const u32x4*)(p + 16));
  return cat8(lo, hi);
}

// ---------------- Tensor Data Mover staging ----------------

#if HAVE_TDM
// 2D bf16 tile load: tile_d0 contiguous elements x tile_d1 rows -> LDS row-major.
// D# packed per CDNA5 ISA 8.3/8.4 (group0: count/lds/global/type, group1: dims).
// This toolchain carries the 6-arg builtin: (g0, g1, g2, g3, g4, cpol).
__device__ __forceinline__ void tdm_load_2d_bf16(const unsigned short* gsrc,
                                                 unsigned lds_off,
                                                 unsigned tensor_d0, unsigned tensor_d1,
                                                 unsigned stride0_elems,
                                                 unsigned tile_d0, unsigned tile_d1) {
  const unsigned long long ga = (unsigned long long)(size_t)gsrc;
  u32x4 g0;
  g0[0] = 1u;                                                  // count=1 (valid D#)
  g0[1] = lds_off;                                             // LDS byte address
  g0[2] = (unsigned)ga;                                        // global_addr[31:0]
  g0[3] = (unsigned)((ga >> 32) & 0x01FFFFFFu) | (2u << 30);   // addr[56:32] | type=2
  i32x8 g1;
  g1[0] = (int)(1u << 16);                                     // data_size = 2 bytes
  g1[1] = (int)((tensor_d0 & 0xFFFFu) << 16);                  // tensor_dim0[15:0]
  g1[2] = (int)(((tensor_d0 >> 16) & 0xFFFFu) | ((tensor_d1 & 0xFFFFu) << 16));
  g1[3] = (int)(((tensor_d1 >> 16) & 0xFFFFu) | (tile_d0 << 16));
  g1[4] = (int)(tile_d1 & 0xFFFFu);                            // tile_dim1 (dim2=0)
  g1[5] = (int)stride0_elems;                                  // dim0_stride[31:0]
  g1[6] = 0;
  g1[7] = 0;
  const i32x4 gz4 = {0, 0, 0, 0};
  const i32x8 gz8 = {0, 0, 0, 0, 0, 0, 0, 0};
  __builtin_amdgcn_tensor_load_to_lds(g0, g1, gz4, gz4, gz8, 0);
}
#endif

// Fallback cooperative stage of a [64 x 32] bf16 tile (256 threads, one b128 each)
__device__ __forceinline__ void stage_fallback(const unsigned short* __restrict__ g,
                                               unsigned short* l, int stride_elems) {
  const int c = threadIdx.x;                  // 0..255 -> 256 x 16B = 4KB
  const int row = c >> 2, off = (c & 3) << 3;
  *(u32x4*)(l + (row << 5) + off) = *(const u32x4*)(g + (size_t)row * stride_elems + off);
}

// Stage a [32 c][64 px] fp32 activation tile into LDS as transposed bf16 [64 px][32 c].
// Each of 256 threads: 8 coalesced b32 loads, 4 packed converts, one ds_store_b128.
__device__ __forceinline__ void stage_x_tile(const float* __restrict__ xb, int c0, int n0,
                                             unsigned short* l) {
  const int t = threadIdx.x;
  const int px = t & 63;                      // pixel within group
  const int c8 = (t >> 6) << 3;               // channel sub-block: 0,8,16,24
  const float* g = xb + (size_t)(c0 + c8) * HW_DIM + n0 + px;
  u32x4 pk;
#pragma unroll
  for (int j = 0; j < 4; ++j)
    pk[j] = pack2bf(g[(size_t)(2 * j) * HW_DIM], g[(size_t)(2 * j + 1) * HW_DIM]);
  *(u32x4*)(l + (px << 5) + c8) = pk;
}

// ---------------- kernels ----------------

__global__ void init_stats_kernel(float* __restrict__ colmax, float* __restrict__ colsum) {
  const int i = blockIdx.x * blockDim.x + threadIdx.x;
  colmax[i] = -3.4e38f;
  colsum[i] = 0.0f;
}

// y = relu(scale*(W @ x) + shift), x: [B, C, HW] fp32, W: [C, C] fp32, y: bf16
// transpose_out: 0 -> y[b, o, p] ([B,C,HW]), 1 -> y[b, p, o] ([B,HW,C])
// Block: 64-pixel group shared by 8 waves; x tile staged to LDS as transposed bf16
// (double-buffered). Wave w computes M tiles w and w+8 (when < 12).
__global__ void proj_gemm_kernel(const float* __restrict__ x, const float* __restrict__ Wm,
                                 const float* __restrict__ bnG, const float* __restrict__ bnB,
                                 const float* __restrict__ bnM, const float* __restrict__ bnV,
                                 unsigned short* __restrict__ y, int transpose_out) {
  __shared__ __align__(16) unsigned short xtile[2][64 * 32];
  const int lane = threadIdx.x & 31;
  const int wid = threadIdx.x >> 5;
  const int bid = blockIdx.x;
  const int b = bid >> 6;                    // 64 n-groups per batch
  const int n0 = (bid & 63) << 6;
  const float* xb = x + (size_t)b * C_DIM * HW_DIM;

  stage_x_tile(xb, 0, n0, &xtile[0][0]);

  v8f zero = {};
  v8f acc[2][4] = {{zero, zero, zero, zero}, {zero, zero, zero, zero}};
  int cur = 0;
#pragma unroll
  for (int c0 = 0; c0 < C_DIM; c0 += 32) {
    if (c0 + 32 < C_DIM) stage_x_tile(xb, c0 + 32, n0, &xtile[cur ^ 1][0]);
    __syncthreads();
#pragma unroll
    for (int s = 0; s < 2; ++s) {
      const int mt = wid + (s << 3);
      if (mt < 12) {
        v16bf A = loadA_f32v(Wm, mt << 4, c0, C_DIM, lane);
#pragma unroll
        for (int t = 0; t < 4; ++t)
          acc[s][t] = wmma_bf16(A, loadB_lds(&xtile[cur][0], t, lane), acc[s][t]);
      }
    }
    __syncthreads();
    cur ^= 1;
  }

  const int nl = lane & 15, mo = (lane >> 4) << 3;
#pragma unroll
  for (int s = 0; s < 2; ++s) {
    const int mt = wid + (s << 3);
    if (mt >= 12) continue;
    const int o0 = mt << 4;
    float sc[8], sh[8];
#pragma unroll
    for (int i = 0; i < 8; ++i) {
      const int o = o0 + mo + i;
      sc[i] = bnG[o] * rsqrtf(bnV[o] + 1e-5f);
      sh[i] = bnB[o] - bnM[o] * sc[i];
    }
    if (transpose_out) {
      unsigned short* yb = y + (size_t)b * HW_DIM * C_DIM;
#pragma unroll
      for (int t = 0; t < 4; ++t) {
        const int p = n0 + (t << 4) + nl;
        u32x4 pk;
#pragma unroll
        for (int j = 0; j < 4; ++j) {
          const float ya = fmaxf(acc[s][t][2 * j] * sc[2 * j] + sh[2 * j], 0.0f);
          const float yb2 =
              fmaxf(acc[s][t][2 * j + 1] * sc[2 * j + 1] + sh[2 * j + 1], 0.0f);
          pk[j] = pack2bf(ya, yb2);
        }
        *(u32x4*)(yb + (size_t)p * C_DIM + o0 + mo) = pk;    // 16B store
      }
    } else {
      unsigned short* yb = y + (size_t)b * C_DIM * HW_DIM;
#pragma unroll
      for (int i = 0; i < 8; ++i) {
        const int o = o0 + mo + i;
#pragma unroll
        for (int t = 0; t < 4; ++t) {
          const float yv = fmaxf(acc[s][t][i] * sc[i] + sh[i], 0.0f);
          yb[(size_t)o * HW_DIM + n0 + (t << 4) + nl] = f2bf(yv);
        }
      }
    }
  }
}

// corr_raw[b,q,k] = (1/sqrt(C)) * sum_c qT[b,q,c]*kT[b,k,c]; fused column-max over q.
// Block: 8 waves share one 64-pixel k-panel (TDM-staged, double buffered in LDS),
// each wave owns one q tile.
__global__ void corr_gemm_kernel(const unsigned short* __restrict__ qT,
                                 const unsigned short* __restrict__ kT,
                                 float* __restrict__ corr, float* __restrict__ colmax) {
  __shared__ __align__(16) unsigned short ktile[2][64 * 32];
  const int lane = threadIdx.x & 31;
  const int wid = threadIdx.x >> 5;
  const int bid = blockIdx.x;
  const int b = bid >> 11;                 // 2048 blocks per batch
  const int r = bid & 2047;
  const int q0 = ((r >> 6) << 7) + (wid << 4);
  const int k0 = (r & 63) << 6;
  const unsigned short* qb = qT + (size_t)b * HW_DIM * C_DIM;
  const unsigned short* kb = kT + (size_t)b * HW_DIM * C_DIM + (size_t)k0 * C_DIM;

#if HAVE_TDM
  if (wid == 0)
    tdm_load_2d_bf16(kb, (unsigned)(size_t)&ktile[0][0], C_DIM, HW_DIM, C_DIM, 32, 64);
#else
  stage_fallback(kb, &ktile[0][0], C_DIM);
#endif

  v8f zero = {};
  v8f acc[4] = {zero, zero, zero, zero};
  int cur = 0;
  for (int c0 = 0; c0 < C_DIM; c0 += 32) {
    const bool more = (c0 + 32 < C_DIM);
#if HAVE_TDM
    if (wid == 0) {
      if (more) {
        tdm_load_2d_bf16(kb + (c0 + 32), (unsigned)(size_t)&ktile[cur ^ 1][0], C_DIM,
                         HW_DIM, C_DIM, 32, 64);
        __builtin_amdgcn_s_wait_tensorcnt(1);
      } else {
        __builtin_amdgcn_s_wait_tensorcnt(0);
      }
    }
#else
    if (more) stage_fallback(kb + (c0 + 32), &ktile[cur ^ 1][0], C_DIM);
#endif
    __syncthreads();
    v16bf A = loadA_bf16v(qb, q0, c0, C_DIM, lane);
#pragma unroll
    for (int t = 0; t < 4; ++t)
      acc[t] = wmma_bf16(A, loadB_lds(&ktile[cur][0], t, lane), acc[t]);
    __syncthreads();
    cur ^= 1;
  }

  const float s = 0.07216878364870323f;    // 192^-0.5
  float* cb = corr + (size_t)b * HW_DIM * HW_DIM;
  const int nl = lane & 15, mo = (lane >> 4) << 3;
#pragma unroll
  for (int t = 0; t < 4; ++t) {
    float mx = -3.4e38f;
#pragma unroll
    for (int i = 0; i < 8; ++i) {
      const float vv = acc[t][i] * s;
      cb[(size_t)(q0 + mo + i) * HW_DIM + (k0 + (t << 4) + nl)] = vv;
      mx = fmaxf(mx, vv);
    }
    mx = fmaxf(mx, __shfl_xor(mx, 16, 32));
    if (lane < 16) atomicMaxF(&colmax[b * HW_DIM + k0 + (t << 4) + nl], mx);
  }
}

// exp(x - colmax) in place + column sum (softmax over q axis); coalesced column scan
__global__ void softmax_expsum_kernel(float* __restrict__ corr,
                                      const float* __restrict__ colmax,
                                      float* __restrict__ colsum) {
  const int t = blockIdx.x * blockDim.x + threadIdx.x;
  const int k = t & (HW_DIM - 1);
  const int r = t >> 12;
  const int sl = r & (NSPLIT - 1);
  const int b = r >> 4;
  float* cb = corr + (size_t)b * HW_DIM * HW_DIM + k;
  const float mx = colmax[b * HW_DIM + k];
  float s = 0.0f;
  const int q0 = sl * (HW_DIM / NSPLIT);
  for (int q = q0; q < q0 + HW_DIM / NSPLIT; ++q) {
    __builtin_prefetch(&cb[(size_t)(q + 8) * HW_DIM], 0, 1);
    const float e = __expf(cb[(size_t)q * HW_DIM] - mx);
    cb[(size_t)q * HW_DIM] = e;
    s += e;
  }
  atomicAdd(&colsum[b * HW_DIM + k], s);
}

__global__ void softmax_norm_kernel(float* __restrict__ corr,
                                    const float* __restrict__ colsum) {
  const size_t total = (size_t)B_DIM * HW_DIM * HW_DIM;
  for (size_t i = (size_t)blockIdx.x * blockDim.x + threadIdx.x; i < total;
       i += (size_t)gridDim.x * blockDim.x) {
    const int k = (int)(i & (HW_DIM - 1));
    const int b = (int)(i >> 24);
    corr[i] = corr[i] / colsum[b * HW_DIM + k];
  }
}

// ss[b,q,c] = sum_k corr[b,q,k] * v[b,c,k]; v stored channel-major [B, C, HW] bf16.
// Block: 8 waves share one 64-channel v-panel per k-step (TDM double buffer).
__global__ void ss_gemm_kernel(const float* __restrict__ corr,
                               const unsigned short* __restrict__ vM,
                               unsigned short* __restrict__ ssb) {
  __shared__ __align__(16) unsigned short vtile[2][64 * 32];
  const int lane = threadIdx.x & 31;
  const int wid = threadIdx.x >> 5;
  const int bid = blockIdx.x;
  const int b = bid / 96;                  // 32 q-blocks x 3 c-groups per batch
  const int r = bid - b * 96;
  const int qblk = r / 3;
  const int cg = r - qblk * 3;
  const int q0 = (qblk << 7) + (wid << 4);
  const int c0b = cg << 6;
  const float* cb = corr + (size_t)b * HW_DIM * HW_DIM;
  const unsigned short* vb = vM + (size_t)b * C_DIM * HW_DIM + (size_t)c0b * HW_DIM;

#if HAVE_TDM
  if (wid == 0)
    tdm_load_2d_bf16(vb, (unsigned)(size_t)&vtile[0][0], HW_DIM, C_DIM, HW_DIM, 32, 64);
#else
  stage_fallback(vb, &vtile[0][0], HW_DIM);
#endif

  v8f zero = {};
  v8f acc[4] = {zero, zero, zero, zero};
  int cur = 0;
  for (int k0 = 0; k0 < HW_DIM; k0 += 32) {
    const bool more = (k0 + 32 < HW_DIM);
#if HAVE_TDM
    if (wid == 0) {
      if (more) {
        tdm_load_2d_bf16(vb + (k0 + 32), (unsigned)(size_t)&vtile[cur ^ 1][0], HW_DIM,
                         C_DIM, HW_DIM, 32, 64);
        __builtin_amdgcn_s_wait_tensorcnt(1);
      } else {
        __builtin_amdgcn_s_wait_tensorcnt(0);
      }
    }
#else
    if (more) stage_fallback(vb + (k0 + 32), &vtile[cur ^ 1][0], HW_DIM);
#endif
    __syncthreads();
    v16bf A = loadA_f32v(cb, q0, k0, HW_DIM, lane);
#pragma unroll
    for (int t = 0; t < 4; ++t)
      acc[t] = wmma_bf16(A, loadB_lds(&vtile[cur][0], t, lane), acc[t]);
    __syncthreads();
    cur ^= 1;
  }

  unsigned short* sb = ssb + (size_t)b * HW_DIM * C_DIM;
  const int nl = lane & 15, mo = (lane >> 4) << 3;
#pragma unroll
  for (int t = 0; t < 4; ++t)
#pragma unroll
    for (int i = 0; i < 8; ++i)
      sb[(size_t)(q0 + mo + i) * C_DIM + (c0b + (t << 4) + nl)] = f2bf(acc[t][i]);
}

// out[b,o,p] = relu(scale*(Wt @ ss^T) + shift), ss: bf16 [B, HW, C], out fp32 [B,C,HW]
__global__ void out_gemm_kernel(const unsigned short* __restrict__ ssb,
                                const float* __restrict__ Wm,
                                const float* __restrict__ bnG, const float* __restrict__ bnB,
                                const float* __restrict__ bnM, const float* __restrict__ bnV,
                                float* __restrict__ out) {
  const int lane = threadIdx.x & 31;
  const int wg = blockIdx.x * WAVES_PER_BLOCK + (threadIdx.x >> 5);
  const int WPB = 12 * 64;
  const int b = wg / WPB;
  const int r = wg - b * WPB;
  const int o0 = (r >> 6) << 4;
  const int n0 = (r & 63) << 6;
  const unsigned short* sb = ssb + (size_t)b * HW_DIM * C_DIM;

  v8f zero = {};
  v8f acc[4] = {zero, zero, zero, zero};
#pragma unroll
  for (int c0 = 0; c0 < C_DIM; c0 += 32) {
    v16bf A = loadA_f32v(Wm, o0, c0, C_DIM, lane);
#pragma unroll
    for (int t = 0; t < 4; ++t)
      acc[t] = wmma_bf16(A, loadBT_bf16v(sb, c0, n0 + (t << 4), C_DIM, lane), acc[t]);
  }
  const int nl = lane & 15, mo = (lane >> 4) << 3;
#pragma unroll
  for (int i = 0; i < 8; ++i) {
    const int o = o0 + mo + i;
    const float sc = bnG[o] * rsqrtf(bnV[o] + 1e-5f);
    const float sh = bnB[o] - bnM[o] * sc;
#pragma unroll
    for (int t = 0; t < 4; ++t)
      out[((size_t)b * C_DIM + o) * HW_DIM + (n0 + (t << 4) + nl)] =
          fmaxf(acc[t][i] * sc + sh, 0.0f);
  }
}

// ---------------- launcher ----------------

extern "C" void kernel_launch(void* const* d_in, const int* in_sizes, int n_in,
                              void* d_out, int out_size, void* d_ws, size_t ws_size,
                              hipStream_t stream) {
  (void)in_sizes; (void)n_in; (void)out_size; (void)ws_size;

  const float* lr = (const float*)d_in[0];
  const float* rf = (const float*)d_in[1];
  const float* Wq = (const float*)d_in[2];
  const float* gq = (const float*)d_in[3];
  const float* bq = (const float*)d_in[4];
  const float* mq = (const float*)d_in[5];
  const float* vq = (const float*)d_in[6];
  const float* Wk = (const float*)d_in[7];
  const float* gk = (const float*)d_in[8];
  const float* bk = (const float*)d_in[9];
  const float* mk = (const float*)d_in[10];
  const float* vk = (const float*)d_in[11];
  const float* Wv = (const float*)d_in[12];
  const float* gv = (const float*)d_in[13];
  const float* bv = (const float*)d_in[14];
  const float* mv = (const float*)d_in[15];
  const float* vv = (const float*)d_in[16];
  const float* Wt = (const float*)d_in[17];
  const float* gt = (const float*)d_in[18];
  const float* bt = (const float*)d_in[19];
  const float* mt = (const float*)d_in[20];
  const float* vt = (const float*)d_in[21];

  float* out  = (float*)d_out;
  float* corr = out + (size_t)B_DIM * C_DIM * HW_DIM;   // (out, corr) concatenated

  const size_t NB = (size_t)B_DIM * HW_DIM * C_DIM;
  unsigned short* qT  = (unsigned short*)d_ws;          // [B, HW, C] bf16
  unsigned short* kT  = qT + NB;                        // [B, HW, C] bf16
  unsigned short* vM  = kT + NB;                        // [B, C, HW] bf16
  unsigned short* ssb = vM + NB;                        // [B, HW, C] bf16
  float* colmax = (float*)(ssb + NB);                   // [B, HW]
  float* colsum = colmax + (size_t)B_DIM * HW_DIM;      // [B, HW]

  init_stats_kernel<<<(B_DIM * HW_DIM) / 256, 256, 0, stream>>>(colmax, colsum);

  // projections: 64 pixel-groups x 4 batches = 256 blocks of 8 waves
  proj_gemm_kernel<<<256, TPB, 0, stream>>>(rf, Wq, gq, bq, mq, vq, qT, 1);
  proj_gemm_kernel<<<256, TPB, 0, stream>>>(lr, Wk, gk, bk, mk, vk, kT, 1);
  proj_gemm_kernel<<<256, TPB, 0, stream>>>(lr, Wv, gv, bv, mv, vv, vM, 0);

  corr_gemm_kernel<<<8192, TPB, 0, stream>>>(qT, kT, corr, colmax);

  softmax_expsum_kernel<<<(B_DIM * NSPLIT * HW_DIM) / 256, 256, 0, stream>>>(corr, colmax,
                                                                             colsum);
  softmax_norm_kernel<<<16384, 256, 0, stream>>>(corr, colsum);

  ss_gemm_kernel<<<384, TPB, 0, stream>>>(corr, vM, ssb);
  out_gemm_kernel<<<384, TPB, 0, stream>>>(ssb, Wt, gt, bt, mt, vt, out);
}
